// Qwen3NextGatedDeltaNet_26654567038981
// MI455X (gfx1250) — matrix operand order repaired and checked by reading, not compiled, not run on hardware
//
#include <hip/hip_runtime.h>
#include <hip/hip_bf16.h>
#include <math.h>

// ---------------------------------------------------------------------------
// Qwen3-Next GatedDeltaNet for MI455X (gfx1250, wave32, WMMA)
//   B=2 S=1024 HIDDEN=2048 NUM_K=16 NUM_V=32 DK=DV=128 KCONV=4
// Pipeline:
//   [prep]  f32 -> bf16 conversions, weight transposes (N x K for B operand)
//   [gemm]  qkvz = hs @ W_qkvz ; ba = hs @ W_ba      (v_wmma_f32_16x16x32_bf16,
//           4x4 register blocking: 16 WMMAs per 16 b128 loads)
//   [conv]  depthwise causal conv(4) + SiLU + per-head L2 norm -> qn,kn,vv
//   [gate]  g = -exp(A_log)*softplus(a+dt_bias) ; beta = sigmoid(b)
//   [scan]  sequential delta rule, 64 blocks, state in VGPRs
//   [norm]  RMS-norm * norm_w * SiLU(z) -> bf16
//   [gemm]  out = core @ W_out                        (WMMA)
// ---------------------------------------------------------------------------

typedef __attribute__((ext_vector_type(16))) __bf16 v16bf;
typedef __attribute__((ext_vector_type(8)))  float  v8f;

#define BB   2
#define SS   1024
#define HID  2048
#define BS   (BB * SS)          // 2048 rows
#define NK   16
#define NV   32
#define DK_  128
#define DV_  128
#define QKVZ_N 12288            // NK * 768
#define MIXC 8192               // 2048 q + 2048 k + 4096 v
#define VALD 4096               // NV*DV

// ------------------------------ prep kernels -------------------------------

__global__ __launch_bounds__(256) void k_cvt_bf16(const float* __restrict__ x,
                                                  __bf16* __restrict__ y, int n) {
  int i = blockIdx.x * 256 + threadIdx.x;
  if (i < n) y[i] = (__bf16)x[i];
}

// W (K x N, f32, row-major)  ->  Wt (N x K, bf16, row-major)
__global__ __launch_bounds__(256) void k_transpose_bf16(const float* __restrict__ W,
                                                        __bf16* __restrict__ Wt,
                                                        int K, int N) {
  int k = blockIdx.x * 32 + (threadIdx.x & 31);
  int n = blockIdx.y * 8 + (threadIdx.x >> 5);
  if (k < K && n < N) Wt[(size_t)n * K + k] = (__bf16)W[(size_t)k * N + n];
}

// ------------------------------ WMMA GEMM ----------------------------------
// C(MxN,f32) = A(MxK,bf16 row-major) * Bt(NxK,bf16 row-major)^T
// block = 128 threads = 4 waves arranged 2x2; each wave computes a 64x64 tile
// with 4x4 register blocking (16 accumulators, 16 WMMAs per k-step of 32).
// A fragment per lane: row m+*(lane&15); chunks at K+8*half, K+16+8*half.
// B fragment per lane: row n+(lane&15) of Bt; chunks at K+8*half, K+16+8*half.
__global__ __launch_bounds__(128)
void k_wmma_gemm(const __bf16* __restrict__ A, const __bf16* __restrict__ Bt,
                 float* __restrict__ C, int M, int N, int K) {
  const int wave = threadIdx.x >> 5;
  const int lane = threadIdx.x & 31;
  const int half = lane >> 4;
  const int l16  = lane & 15;
  const int m0 = blockIdx.y * 128 + (wave >> 1) * 64;   // wave's 64-row strip
  const int n0 = blockIdx.x * 128 + (wave & 1) * 64;    // wave's 64-col strip
  if (m0 >= M || n0 >= N) return;

  v8f acc[4][4];
#pragma unroll
  for (int i = 0; i < 4; ++i)
#pragma unroll
    for (int j = 0; j < 4; ++j) acc[i][j] = (v8f){};

  const __bf16* arow0 = A + (size_t)(m0 + l16) * K + half * 8;
  const __bf16* brow0 = Bt + (size_t)(n0 + l16) * K + half * 8;

  for (int k = 0; k < K; k += 32) {
    v16bf af[4];
#pragma unroll
    for (int i = 0; i < 4; ++i) {
      const __bf16* ap = arow0 + (size_t)i * 16 * K + k;
      ((uint4*)&af[i])[0] = *(const uint4*)(ap);        // K = k+8h+0..7
      ((uint4*)&af[i])[1] = *(const uint4*)(ap + 16);   // K = k+16+8h+0..7
    }
#pragma unroll
    for (int j = 0; j < 4; ++j) {
      const __bf16* bp = brow0 + (size_t)j * 16 * K + k;
      v16bf b;
      ((uint4*)&b)[0] = *(const uint4*)(bp);            // K = k+8h+0..7
      ((uint4*)&b)[1] = *(const uint4*)(bp + 16);       // K = k+16+8h+0..7
#pragma unroll
      for (int i = 0; i < 4; ++i)
        acc[i][j] = __builtin_amdgcn_wmma_f32_16x16x32_bf16(
            false, af[i], false, b, (short)0, acc[i][j], false, false);
    }
  }
#pragma unroll
  for (int i = 0; i < 4; ++i)
#pragma unroll
    for (int j = 0; j < 4; ++j)
#pragma unroll
      for (int r = 0; r < 8; ++r)
        C[(size_t)(m0 + i * 16 + half * 8 + r) * N + n0 + j * 16 + l16] =
            acc[i][j][r];
}

// --------------------- conv + SiLU + per-head L2 norm ----------------------
__global__ __launch_bounds__(256)
void k_conv_norm(const float* __restrict__ qkvz, const float* __restrict__ cw,
                 float* __restrict__ qn, float* __restrict__ kn,
                 float* __restrict__ vv) {
  const int bs = blockIdx.x;          // b*S + s
  const int b = bs / SS, s = bs % SS;
  __shared__ float sm[MIXC];
  __shared__ float rsq[32];

  for (int c = threadIdx.x; c < MIXC; c += 256) {
    int col;
    if (c < 2048)       { col = (c >> 7) * 768 + (c & 127); }
    else if (c < 4096)  { int c2 = c - 2048; col = (c2 >> 7) * 768 + 128 + (c2 & 127); }
    else                { int c3 = c - 4096; col = (c3 >> 8) * 768 + 256 + (c3 & 255); }
    float acc = 0.f;
#pragma unroll
    for (int t = 0; t < 4; ++t) {
      int sp = s - 3 + t;
      float xv = (sp >= 0) ? qkvz[((size_t)b * SS + sp) * QKVZ_N + col] : 0.f;
      acc += xv * cw[c * 4 + t];
    }
    sm[c] = acc / (1.f + __expf(-acc));   // SiLU
  }
  __syncthreads();
  if (threadIdx.x < 32) {                 // groups: 0..15 q heads, 16..31 k heads
    int base = threadIdx.x * 128;
    float ssum = 0.f;
    for (int d = 0; d < 128; ++d) { float xv = sm[base + d]; ssum += xv * xv; }
    rsq[threadIdx.x] = rsqrtf(ssum + 1e-6f);
  }
  __syncthreads();
  const float qscale = 0.08838834764831845f;   // 128^-0.5
  for (int c = threadIdx.x; c < 2048; c += 256) {
    int h = c >> 7, d = c & 127;
    qn[((size_t)bs * NK + h) * DK_ + d] = sm[c] * rsq[h] * qscale;
    kn[((size_t)bs * NK + h) * DK_ + d] = sm[2048 + c] * rsq[16 + h];
  }
  for (int c = threadIdx.x; c < VALD; c += 256)
    vv[(size_t)bs * VALD + c] = sm[4096 + c];
}

// ------------------------------ g / beta -----------------------------------
__global__ __launch_bounds__(256)
void k_gbeta(const float* __restrict__ ba, const float* __restrict__ A_log,
             const float* __restrict__ dt_bias, float* __restrict__ g,
             float* __restrict__ beta, int total) {
  int i = blockIdx.x * 256 + threadIdx.x;
  if (i >= total) return;
  int vh = i & 31, nk = vh >> 1, r = vh & 1;
  size_t bs = (size_t)(i >> 5);
  float bv = ba[bs * 64 + nk * 4 + r];
  float av = ba[bs * 64 + nk * 4 + 2 + r];
  float x = av + dt_bias[vh];
  float sp = (x > 20.f) ? x : log1pf(__expf(x));
  g[i] = -__expf(A_log[vh]) * sp;
  beta[i] = 1.f / (1.f + __expf(-bv));
}

// --------------------------- sequential scan -------------------------------
// 64 blocks (b*32+vh), 256 threads. Thread (v0 = tid&127, rh = tid>>7) owns
// state rows [rh*64, rh*64+64) of column v0, fully in VGPRs.
__global__ __launch_bounds__(256)
void k_scan(const float* __restrict__ qn, const float* __restrict__ kn,
            const float* __restrict__ vv, const float* __restrict__ g,
            const float* __restrict__ beta, float* __restrict__ core) {
  const int b  = blockIdx.x >> 5;
  const int vh = blockIdx.x & 31;
  const int kh = vh >> 1;
  const int tid = threadIdx.x;
  const int v0 = tid & 127;
  const int rh = tid >> 7;
  __shared__ float sk[128], sq[128], sv[128], sdelta[128], red[256], sgb[2];
  float st[64];
#pragma unroll
  for (int j = 0; j < 64; ++j) st[j] = 0.f;

  for (int t = 0; t < SS; ++t) {
    const size_t bs = (size_t)b * SS + t;
    if (tid < 128) {
      sk[tid] = kn[(bs * NK + kh) * DK_ + tid];
      sq[tid] = qn[(bs * NK + kh) * DK_ + tid];
      sv[tid] = vv[(bs * NV + vh) * DV_ + tid];
    } else if (tid == 128) {
      sgb[0] = __expf(g[bs * NV + vh]);
      sgb[1] = beta[bs * NV + vh];
    }
    __syncthreads();
    const float eg = sgb[0], bt = sgb[1];
    const float* krow = &sk[rh * 64];
    float p = 0.f;
#pragma unroll
    for (int j = 0; j < 64; ++j) p += krow[j] * st[j];
    red[tid] = p;
    __syncthreads();
    if (tid < 128) {
      float pred = (red[tid] + red[tid + 128]) * eg;
      sdelta[tid] = (sv[tid] - pred) * bt;
    }
    __syncthreads();
    const float dl = sdelta[v0];
    const float* qrow = &sq[rh * 64];
    float o = 0.f;
#pragma unroll
    for (int j = 0; j < 64; ++j) {
      float sn = st[j] * eg + krow[j] * dl;
      st[j] = sn;
      o += qrow[j] * sn;
    }
    red[tid] = o;
    __syncthreads();
    if (tid < 128) core[(bs * NV + vh) * DV_ + tid] = red[tid] + red[tid + 128];
    __syncthreads();
  }
}

// ------------------- RMS norm * norm_w * SiLU(z) -> bf16 -------------------
__global__ __launch_bounds__(128)
void k_norm_gate(const float* __restrict__ core, const float* __restrict__ qkvz,
                 const float* __restrict__ nw, __bf16* __restrict__ coreb) {
  const int i = blockIdx.x;                 // bs*32 + vh
  const int d = threadIdx.x;
  const int vh = i & 31, nk = vh >> 1, r = vh & 1;
  const size_t bs = (size_t)(i >> 5);
  float x = core[(size_t)i * DV_ + d];
  __shared__ float red[128];
  red[d] = x * x;
  __syncthreads();
  for (int off = 64; off > 0; off >>= 1) {
    if (d < off) red[d] += red[d + off];
    __syncthreads();
  }
  float rms = rsqrtf(red[0] * (1.f / 128.f) + 1e-6f);
  float z = qkvz[bs * QKVZ_N + nk * 768 + 512 + r * 128 + d];
  float sz = z / (1.f + __expf(-z));
  coreb[bs * VALD + (size_t)vh * DV_ + d] = (__bf16)(x * rms * nw[d] * sz);
}

// ------------------------------- launcher ----------------------------------

extern "C" void kernel_launch(void* const* d_in, const int* in_sizes, int n_in,
                              void* d_out, int out_size, void* d_ws, size_t ws_size,
                              hipStream_t stream) {
  const float* hs      = (const float*)d_in[0];   // (B,S,HID)
  const float* W_qkvz  = (const float*)d_in[1];   // (HID, 12288)
  const float* W_ba    = (const float*)d_in[2];   // (HID, 64)
  const float* conv_w  = (const float*)d_in[3];   // (8192, 4)
  const float* A_log   = (const float*)d_in[4];   // (32,)
  const float* dt_bias = (const float*)d_in[5];   // (32,)
  const float* norm_w  = (const float*)d_in[6];   // (128,)
  const float* W_out   = (const float*)d_in[7];   // (4096, HID)
  float* out = (float*)d_out;

  // workspace carve-out (256B aligned slices)
  size_t off = 0;
  auto carve = [&](size_t bytes) {
    void* p = (char*)d_ws + off;
    off = (off + bytes + 255) & ~(size_t)255;
    return p;
  };
  __bf16* hsb     = (__bf16*)carve((size_t)BS * HID * 2);
  __bf16* wqkvzT  = (__bf16*)carve((size_t)QKVZ_N * HID * 2);
  __bf16* wbaT    = (__bf16*)carve((size_t)64 * HID * 2);
  __bf16* woutT   = (__bf16*)carve((size_t)HID * VALD * 2);
  float*  qkvz    = (float*) carve((size_t)BS * QKVZ_N * 4);
  float*  ba      = (float*) carve((size_t)BS * 64 * 4);
  float*  qn      = (float*) carve((size_t)BS * NK * DK_ * 4);
  float*  kn      = (float*) carve((size_t)BS * NK * DK_ * 4);
  float*  vv      = (float*) carve((size_t)BS * VALD * 4);
  float*  gbuf    = (float*) carve((size_t)BS * NV * 4);
  float*  bbuf    = (float*) carve((size_t)BS * NV * 4);
  float*  core    = (float*) carve((size_t)BS * VALD * 4);
  __bf16* coreb   = (__bf16*)carve((size_t)BS * VALD * 2);
  if (off > ws_size) return;   // workspace too small: bail deterministically

  // prep: conversions + transposes
  {
    int n = BS * HID;
    k_cvt_bf16<<<(n + 255) / 256, 256, 0, stream>>>(hs, hsb, n);
  }
  k_transpose_bf16<<<dim3(HID / 32, QKVZ_N / 8), 256, 0, stream>>>(W_qkvz, wqkvzT, HID, QKVZ_N);
  k_transpose_bf16<<<dim3(HID / 32, 64 / 8), 256, 0, stream>>>(W_ba, wbaT, HID, 64);
  k_transpose_bf16<<<dim3(VALD / 32, HID / 8), 256, 0, stream>>>(W_out, woutT, VALD, HID);

  // qkvz & ba GEMMs (block tiles 128x128; ba's N=64 handled by n0>=N guard)
  k_wmma_gemm<<<dim3(QKVZ_N / 128, BS / 128), 128, 0, stream>>>(hsb, wqkvzT, qkvz, BS, QKVZ_N, HID);
  k_wmma_gemm<<<dim3(1, BS / 128), 128, 0, stream>>>(hsb, wbaT, ba, BS, 64, HID);

  // conv + SiLU + L2 norm ; gating
  k_conv_norm<<<BS, 256, 0, stream>>>(qkvz, conv_w, qn, kn, vv);
  k_gbeta<<<(BS * NV + 255) / 256, 256, 0, stream>>>(ba, A_log, dt_bias, gbuf, bbuf, BS * NV);

  // sequential delta-rule scan
  k_scan<<<BB * NV, 256, 0, stream>>>(qn, kn, vv, gbuf, bbuf, core);

  // RMS norm + gate -> bf16, then output GEMM
  k_norm_gate<<<BS * NV, 128, 0, stream>>>(core, qkvz, norm_w, coreb);
  k_wmma_gemm<<<dim3(HID / 128, BS / 128), 128, 0, stream>>>(coreb, woutT, out, BS, HID, VALD);
}
